// FeatureAttentionModule_84378927497484
// MI455X (gfx1250) — compile-verified, matrix operand order checked by reading, not run on hardware
//
#include <hip/hip_runtime.h>
#include <hip/hip_bf16.h>

typedef __attribute__((ext_vector_type(16))) _Float16 v16h;
typedef __attribute__((ext_vector_type(8)))  float    v8f;
typedef __attribute__((ext_vector_type(4)))  unsigned int u32x4;
typedef __attribute__((ext_vector_type(8)))  int      i32x8;
typedef __attribute__((ext_vector_type(4)))  int      i32x4;

namespace fam {
constexpr int E    = 128;
constexpr int H    = 4;
constexpr int HD   = 32;    // E / H
constexpr int FF   = 256;
constexpr int NEXP = 5;
constexpr int NL   = 6;
constexpr int B    = 512;
constexpr int L    = 51;
constexpr int NCAT = 53;
constexpr int NNUM = 47;
constexpr int NENG = 100;
constexpr int NTOK = 201;          // 53+47+100+1
constexpr int NPAD = 224;          // 14 tiles of 16; 7 chunks of 32
constexpr int MROWS = B * NPAD;    // padded token rows
constexpr int MT   = MROWS / 16;   // 7168 M-tiles
}

using namespace fam;

// ---------------------------------------------------------------------------
// WMMA helpers. 16-bit operand layout per CDNA5 ISA 7.12.2:
// lane = r + 16*g; VGPR i holds K pair (i<4 ? 2i : 16+2(i-4)) + 8g ->
// two 16-byte loads per lane.
// ---------------------------------------------------------------------------
__device__ __forceinline__ v16h load_tile16(const _Float16* base, int ldk) {
  int lane = threadIdx.x & 31;
  int r = lane & 15, g = lane >> 4;
  union { v16h v; uint4 q[2]; } t;
  const _Float16* row = base + (size_t)r * ldk + 8 * g;
  t.q[0] = *(const uint4*)(row);
  t.q[1] = *(const uint4*)(row + 16);
  return t.v;
}

__device__ __forceinline__ v8f wmma16(v16h a, v16h b, v8f c) {
  return __builtin_amdgcn_wmma_f32_16x16x32_f16(
      false, a, false, b, (short)0, c, false, false);
}

// C/D layout: element (M = i + 8*(lane>>4), N = lane&15) lives in VGPR i.

// ---------------------------------------------------------------------------
// TDM: stage a 2-D f16 tile (rows x cols elems, row-stride in elems) from
// global into LDS. D# built per CDNA5 ISA ch.8 (group0/group1 bitfields,
// data_size=1 (2 bytes), type=2, 2-D -> groups 2/3 zero). Issued by one wave;
// caller barriers afterwards. This toolchain exposes the 6-arg builtin:
// (u32x4 g0, i32x8 g1, i32x4 g2, i32x4 g3, i32x8 extra, i32 cpol).
// ---------------------------------------------------------------------------
#if __has_builtin(__builtin_amdgcn_tensor_load_to_lds) && defined(__gfx1250__)
#define FAM_HAS_TDM 1
#else
#define FAM_HAS_TDM 0
#endif

#if FAM_HAS_TDM
__device__ __forceinline__ void tdm_load_2d(const _Float16* gsrc, unsigned lds_byte,
                                            int rows, int cols, int row_stride) {
  unsigned long long ga = (unsigned long long)(uintptr_t)gsrc;
  u32x4 g0;
  g0[0] = 1u;                                             // count=1 (valid), user mode
  g0[1] = lds_byte;                                       // lds_addr
  g0[2] = (unsigned)(ga & 0xFFFFFFFFu);                   // global_addr[31:0]
  g0[3] = (unsigned)((ga >> 32) & 0x1FFFFFFu) | (2u << 30); // addr[56:32] | type=2
  i32x8 g1;
  g1[0] = (int)(1u << 16);                                // data_size=1 (2B), mask=0
  g1[1] = (int)(((unsigned)cols & 0xFFFFu) << 16);        // tensor_dim0[15:0] @bits63:48
  g1[2] = (int)((((unsigned)cols >> 16) & 0xFFFFu) |      // tensor_dim0[31:16]
                (((unsigned)rows & 0xFFFFu) << 16));      // tensor_dim1[15:0]
  g1[3] = (int)((((unsigned)rows >> 16) & 0xFFFFu) |      // tensor_dim1[31:16]
                (((unsigned)cols & 0xFFFFu) << 16));      // tile_dim0
  g1[4] = (int)((unsigned)rows & 0xFFFFu);                // tile_dim1, tile_dim2=0
  g1[5] = (int)(unsigned)row_stride;                      // tensor_dim0_stride[31:0]
  g1[6] = 0;                                              // stride0 hi, stride1 lo
  g1[7] = 0;
  i32x4 z4 = {0, 0, 0, 0};
  i32x8 z8 = {0, 0, 0, 0, 0, 0, 0, 0};
  __builtin_amdgcn_tensor_load_to_lds(g0, g1, z4, z4, z8, 0);
  __builtin_amdgcn_s_wait_tensorcnt(0);
}
#endif

// stage `rows x 128` f16 weight tile into LDS (all threads participate in
// fallback; only wave 0 issues on the TDM path).
template <int ROWS>
__device__ __forceinline__ void stage_weights(const _Float16* gsrc, _Float16* sW) {
#if FAM_HAS_TDM
  if (threadIdx.x < 32)
    tdm_load_2d(gsrc, (unsigned)(uintptr_t)sW, ROWS, E, E);
#else
  const int nvec = ROWS * E / 8;   // uint4 elems
  for (int idx = threadIdx.x; idx < nvec; idx += (int)blockDim.x)
    ((uint4*)sW)[idx] = ((const uint4*)gsrc)[idx];
#endif
  __syncthreads();
}

// ---------------------------------------------------------------------------
// Weight convert: f32 -> f16; W1 (x,e,f)->(x,f,e), W2 (x,f,e)->(x,e,f) so the
// B operand of every GEMM is K-contiguous per output feature.
// ---------------------------------------------------------------------------
__global__ void k_convert(const float* __restrict__ Wqkv, const float* __restrict__ Wo,
                          const float* __restrict__ W1,   const float* __restrict__ W2,
                          _Float16* __restrict__ Wqkv16, _Float16* __restrict__ Wo16,
                          _Float16* __restrict__ W1t,    _Float16* __restrict__ W2t) {
  int i = blockIdx.x * blockDim.x + threadIdx.x;
  const int n_qkv = NL * 3 * E * E;
  const int n_wo  = NL * E * E;
  const int n_w1  = NL * NEXP * E * FF;
  if (i < n_qkv) { Wqkv16[i] = (_Float16)Wqkv[i]; return; }
  i -= n_qkv;
  if (i < n_wo)  { Wo16[i] = (_Float16)Wo[i]; return; }
  i -= n_wo;
  if (i < n_w1) {
    int le = i / (E * FF), r = i % (E * FF);
    int e = r / FF, f = r % FF;
    W1t[((size_t)le * FF + f) * E + e] = (_Float16)W1[i];
    return;
  }
  i -= n_w1;
  if (i < n_w1) {
    int le = i / (FF * E), r = i % (FF * E);
    int f = r / E, e = r % E;
    W2t[((size_t)le * E + e) * FF + f] = (_Float16)W2[i];
  }
}

// ---------------------------------------------------------------------------
// Embedding + LayerNorm0 -> x (f32). One wave per token row.
// ---------------------------------------------------------------------------
__global__ void k_embed_ln0(const int* __restrict__ x_cat, const float* __restrict__ x_num,
                            const float* __restrict__ x_eng, const float* __restrict__ emb,
                            const float* __restrict__ emb_bias, const float* __restrict__ emb_eng,
                            const float* __restrict__ emb_bias_eng,
                            const float* __restrict__ w, const float* __restrict__ bn,
                            float* __restrict__ x) {
  int row = (int)((blockIdx.x * (unsigned)blockDim.x + threadIdx.x) >> 5);
  if (row >= MROWS) return;
  int lane = threadIdx.x & 31;
  int b = row / NPAD, n = row % NPAD;
  float v[4];
#pragma unroll
  for (int j = 0; j < 4; ++j) {
    int e = lane + 32 * j;
    float val = 0.f;
    if (n < NCAT) {
      int idx = x_cat[(b * L + (L - 1)) * NCAT + n];
      val = emb[(size_t)idx * E + e] + emb_bias[n * E + e];
    } else if (n < NCAT + NNUM) {
      int i2 = n - NCAT;
      val = emb[(size_t)(1306 + i2) * E + e] * x_num[(b * L + (L - 1)) * NNUM + i2]
            + emb_bias[n * E + e];
    } else if (n < 200) {
      int i2 = n - 100;
      val = emb_eng[i2 * E + e] * x_eng[b * NENG + i2] + emb_bias_eng[i2 * E + e];
    }
    v[j] = val;
  }
  float s = v[0] + v[1] + v[2] + v[3];
#pragma unroll
  for (int o = 16; o > 0; o >>= 1) s += __shfl_xor(s, o, 32);
  float m = s * (1.0f / E);
  float q = 0.f;
#pragma unroll
  for (int j = 0; j < 4; ++j) { v[j] -= m; q += v[j] * v[j]; }
#pragma unroll
  for (int o = 16; o > 0; o >>= 1) q += __shfl_xor(q, o, 32);
  float inv = rsqrtf(q * (1.0f / E) + 1e-5f);
  float* xr = x + (size_t)row * E;
#pragma unroll
  for (int j = 0; j < 4; ++j) {
    int e = lane + 32 * j;
    xr[e] = v[j] * inv * w[e] + bn[e];
  }
}

// ---------------------------------------------------------------------------
// LayerNorm (f32 in) -> f16 out. One wave per row.
// ---------------------------------------------------------------------------
__global__ void k_ln_f16(const float* __restrict__ x, const float* __restrict__ w,
                         const float* __restrict__ bn, _Float16* __restrict__ h16) {
  int row = (int)((blockIdx.x * (unsigned)blockDim.x + threadIdx.x) >> 5);
  if (row >= MROWS) return;
  int lane = threadIdx.x & 31;
  const float* xr = x + (size_t)row * E;
  float v[4];
#pragma unroll
  for (int j = 0; j < 4; ++j) v[j] = xr[lane + 32 * j];
  float s = v[0] + v[1] + v[2] + v[3];
#pragma unroll
  for (int o = 16; o > 0; o >>= 1) s += __shfl_xor(s, o, 32);
  float m = s * (1.0f / E);
  float q = 0.f;
#pragma unroll
  for (int j = 0; j < 4; ++j) { v[j] -= m; q += v[j] * v[j]; }
#pragma unroll
  for (int o = 16; o > 0; o >>= 1) q += __shfl_xor(q, o, 32);
  float inv = rsqrtf(q * (1.0f / E) + 1e-5f);
  _Float16* hr = h16 + (size_t)row * E;
#pragma unroll
  for (int j = 0; j < 4; ++j) {
    int e = lane + 32 * j;
    hr[e] = (_Float16)(v[j] * inv * w[e] + bn[e]);
  }
}

// ---------------------------------------------------------------------------
// QKV GEMM with LDS-staged weights (TDM). Block (8 waves) = 8 M-tiles x one
// 64-col N-group; the 64x128 weight tile is loaded once into LDS.
// ---------------------------------------------------------------------------
__global__ void k_gemm_qkv(const _Float16* __restrict__ h16,
                           const _Float16* __restrict__ w16,  // (384 x 128)
                           const float* __restrict__ bias,    // 384
                           _Float16* __restrict__ out) {      // (M x 384)
  __shared__ alignas(16) _Float16 sW[64 * E];                 // 16 KB
  int ng = blockIdx.x % 6;
  int mg = blockIdx.x / 6;
  stage_weights<64>(w16 + (size_t)ng * 64 * E, sW);

  int wv = threadIdx.x >> 5, lane = threadIdx.x & 31;
  int cn = lane & 15, g = lane >> 4;
  int mt = mg * 8 + wv;
  v8f acc[4];
#pragma unroll
  for (int j = 0; j < 4; ++j) {
    float bv = bias[ng * 64 + j * 16 + cn];
#pragma unroll
    for (int i = 0; i < 8; ++i) acc[j][i] = bv;
  }
#pragma unroll
  for (int k = 0; k < 4; ++k) {
    v16h a = load_tile16(h16 + (size_t)mt * 16 * E + k * 32, E);
#pragma unroll
    for (int j = 0; j < 4; ++j) {
      v16h bm = load_tile16(sW + (j * 16) * E + k * 32, E);   // LDS
      acc[j] = wmma16(a, bm, acc[j]);
    }
  }
#pragma unroll
  for (int j = 0; j < 4; ++j)
#pragma unroll
    for (int i = 0; i < 8; ++i) {
      int r = mt * 16 + i + 8 * g;
      out[(size_t)r * 384 + ng * 64 + j * 16 + cn] = (_Float16)acc[j][i];
    }
}

// ---------------------------------------------------------------------------
// Fused attention: one workgroup (4 waves) per (b, head, 16-row q-tile).
// ---------------------------------------------------------------------------
__global__ void k_attn(const _Float16* __restrict__ qkv, _Float16* __restrict__ o16) {
  __shared__ alignas(16) float    sc[16 * NPAD];
  __shared__ alignas(16) _Float16 at[16 * NPAD];
  __shared__ alignas(16) _Float16 vT[HD * NPAD];
  int bid = blockIdx.x;
  int qt = bid % 14; int t2 = bid / 14;
  int hh = t2 % H;   int b  = t2 / H;
  const _Float16* base = qkv + (size_t)b * NPAD * 384;
  const _Float16* qh = base + hh * HD;
  const _Float16* kh = base + E + hh * HD;
  const _Float16* vh = base + 2 * E + hh * HD;
  int wv = threadIdx.x >> 5, lane = threadIdx.x & 31;
  int cn = lane & 15, g = lane >> 4;

  for (int idx = threadIdx.x; idx < HD * NPAD; idx += 128) {
    int c = idx / NPAD, n = idx % NPAD;
    vT[c * NPAD + n] = vh[(size_t)n * 384 + c];
  }
  const float scale = 0.17677669529663689f;  // 1/sqrt(32)
  v16h a = load_tile16(qh + (size_t)qt * 16 * 384, 384);
  for (int nt = wv; nt < 14; nt += 4) {
    v16h bm = load_tile16(kh + (size_t)nt * 16 * 384, 384);
    v8f c = {};
    c = wmma16(a, bm, c);
#pragma unroll
    for (int i = 0; i < 8; ++i) {
      int col = nt * 16 + cn;
      sc[(i + 8 * g) * NPAD + col] = (col < NTOK) ? c[i] * scale : -1e30f;
    }
  }
  __syncthreads();
  if (threadIdx.x < 16) {
    int r = threadIdx.x;
    float mx = -1e30f;
    for (int c0 = 0; c0 < NTOK; ++c0) mx = fmaxf(mx, sc[r * NPAD + c0]);
    float s = 0.f;
    for (int c0 = 0; c0 < NPAD; ++c0) {
      float ev = (c0 < NTOK) ? __expf(sc[r * NPAD + c0] - mx) : 0.f;
      sc[r * NPAD + c0] = ev; s += ev;
    }
    float inv = 1.f / s;
    for (int c0 = 0; c0 < NPAD; ++c0)
      at[r * NPAD + c0] = (_Float16)(sc[r * NPAD + c0] * inv);
  }
  __syncthreads();
  if (wv < 2) {
    v8f c = {};
#pragma unroll
    for (int k = 0; k < 7; ++k) {
      v16h aa = load_tile16(at + k * 32, NPAD);
      v16h bb = load_tile16(vT + (size_t)(wv * 16) * NPAD + k * 32, NPAD);
      c = wmma16(aa, bb, c);
    }
#pragma unroll
    for (int i = 0; i < 8; ++i) {
      int r = qt * 16 + i + 8 * g;
      o16[((size_t)b * NPAD + r) * E + hh * HD + wv * 16 + cn] = (_Float16)c[i];
    }
  }
}

// ---------------------------------------------------------------------------
// Wo GEMM + residual, LDS-staged weights: x += o16 @ Wo^T + bo.
// Block = 8 M-tiles x one 64-col N-group.
// ---------------------------------------------------------------------------
__global__ void k_gemm_wo(const _Float16* __restrict__ o16,
                          const _Float16* __restrict__ w16,  // (128 x 128)
                          const float* __restrict__ bias,    // 128
                          float* __restrict__ x) {
  __shared__ alignas(16) _Float16 sW[64 * E];                 // 16 KB
  int ng = blockIdx.x & 1;
  int mg = blockIdx.x >> 1;
  stage_weights<64>(w16 + (size_t)ng * 64 * E, sW);

  int wv = threadIdx.x >> 5, lane = threadIdx.x & 31;
  int cn = lane & 15, g = lane >> 4;
  int mt = mg * 8 + wv;
  v8f acc[4];
#pragma unroll
  for (int j = 0; j < 4; ++j) {
    float bv = bias[ng * 64 + j * 16 + cn];
#pragma unroll
    for (int i = 0; i < 8; ++i) acc[j][i] = bv;
  }
#pragma unroll
  for (int k = 0; k < 4; ++k) {
    v16h a = load_tile16(o16 + (size_t)mt * 16 * E + k * 32, E);
#pragma unroll
    for (int j = 0; j < 4; ++j) {
      v16h bm = load_tile16(sW + (j * 16) * E + k * 32, E);   // LDS
      acc[j] = wmma16(a, bm, acc[j]);
    }
  }
#pragma unroll
  for (int j = 0; j < 4; ++j)
#pragma unroll
    for (int i = 0; i < 8; ++i) {
      size_t idx = (size_t)(mt * 16 + i + 8 * g) * E + ng * 64 + j * 16 + cn;
      x[idx] += acc[j][i];
    }
}

// ---------------------------------------------------------------------------
// Gate: softmax(h @ Wg + bg) over 5 experts. One thread per token.
// ---------------------------------------------------------------------------
__global__ void k_gate(const _Float16* __restrict__ h16, const float* __restrict__ Wg,
                       const float* __restrict__ bg, float* __restrict__ gate) {
  int t = blockIdx.x * blockDim.x + threadIdx.x;
  if (t >= MROWS) return;
  const _Float16* hr = h16 + (size_t)t * E;
  float lg[NEXP];
#pragma unroll
  for (int xk = 0; xk < NEXP; ++xk) lg[xk] = bg[xk];
  for (int e = 0; e < E; ++e) {
    float hv = (float)hr[e];
#pragma unroll
    for (int xk = 0; xk < NEXP; ++xk) lg[xk] += hv * Wg[e * NEXP + xk];
  }
  float mx = lg[0];
#pragma unroll
  for (int xk = 1; xk < NEXP; ++xk) mx = fmaxf(mx, lg[xk]);
  float s = 0.f;
#pragma unroll
  for (int xk = 0; xk < NEXP; ++xk) { lg[xk] = __expf(lg[xk] - mx); s += lg[xk]; }
  float inv = 1.f / s;
#pragma unroll
  for (int xk = 0; xk < NEXP; ++xk) gate[(size_t)t * NEXP + xk] = lg[xk] * inv;
}

// ---------------------------------------------------------------------------
// Fused MoE per 16-token tile (8 waves): hid = relu(h@W1t^T + b1) -> LDS,
// eo = hid@W2t^T + b2, acc += gate[:,e] * eo; x += acc.
// ---------------------------------------------------------------------------
__global__ void k_moe(const _Float16* __restrict__ h16,
                      const _Float16* __restrict__ W1t,  // (NEXP, FF, E)
                      const _Float16* __restrict__ W2t,  // (NEXP, E, FF)
                      const float* __restrict__ b1,      // (NEXP, FF)
                      const float* __restrict__ b2,      // (NEXP, E)
                      const float* __restrict__ gate,    // (MROWS, NEXP)
                      float* __restrict__ x) {
  __shared__ alignas(16) _Float16 hid[16 * FF];          // 8 KB
  int tile = blockIdx.x;
  int wv = threadIdx.x >> 5, lane = threadIdx.x & 31;
  int cn = lane & 15, g = lane >> 4;
  const _Float16* A0 = h16 + (size_t)tile * 16 * E;
  v8f acc = {};
  for (int e = 0; e < NEXP; ++e) {
    const _Float16* w1 = W1t + (size_t)e * FF * E;
    const float* bb1 = b1 + e * FF;
#pragma unroll
    for (int half = 0; half < 2; ++half) {
      int nt = wv + 8 * half;
      v8f c;
      float bv = bb1[nt * 16 + cn];
#pragma unroll
      for (int i = 0; i < 8; ++i) c[i] = bv;
#pragma unroll
      for (int k = 0; k < 4; ++k) {
        v16h a = load_tile16(A0 + k * 32, E);
        v16h bm = load_tile16(w1 + (size_t)(nt * 16) * E + k * 32, E);
        c = wmma16(a, bm, c);
      }
#pragma unroll
      for (int i = 0; i < 8; ++i)
        hid[(i + 8 * g) * FF + nt * 16 + cn] = (_Float16)fmaxf(c[i], 0.f);
    }
    __syncthreads();
    const _Float16* w2 = W2t + (size_t)e * E * FF;
    v8f c2;
    float bv2 = b2[e * E + wv * 16 + cn];
#pragma unroll
    for (int i = 0; i < 8; ++i) c2[i] = bv2;
#pragma unroll
    for (int k = 0; k < 8; ++k) {
      v16h a = load_tile16(hid + k * 32, FF);
      v16h bm = load_tile16(w2 + (size_t)(wv * 16) * FF + k * 32, FF);
      c2 = wmma16(a, bm, c2);
    }
#pragma unroll
    for (int i = 0; i < 8; ++i) {
      int tok = tile * 16 + i + 8 * g;
      acc[i] += gate[(size_t)tok * NEXP + e] * c2[i];
    }
    __syncthreads();
  }
#pragma unroll
  for (int i = 0; i < 8; ++i) {
    int tok = tile * 16 + i + 8 * g;
    x[(size_t)tok * E + wv * 16 + cn] += acc[i];
  }
}

// ---------------------------------------------------------------------------
__global__ void k_out(const float* __restrict__ x, float* __restrict__ out) {
  int i = blockIdx.x * blockDim.x + threadIdx.x;
  if (i >= B * E) return;
  int b = i / E, e = i % E;
  out[i] = x[((size_t)b * NPAD + 200) * E + e];
}

// ---------------------------------------------------------------------------
extern "C" void kernel_launch(void* const* d_in, const int* in_sizes, int n_in,
                              void* d_out, int out_size, void* d_ws, size_t ws_size,
                              hipStream_t stream) {
  (void)in_sizes; (void)n_in; (void)out_size; (void)ws_size;
  const int*   x_cat       = (const int*)  d_in[0];
  const float* x_num       = (const float*)d_in[1];
  const float* x_eng       = (const float*)d_in[2];
  const float* emb         = (const float*)d_in[3];
  const float* emb_bias    = (const float*)d_in[4];
  const float* emb_eng     = (const float*)d_in[5];
  const float* emb_bias_en = (const float*)d_in[6];
  const float* ln0_w       = (const float*)d_in[7];
  const float* ln0_b       = (const float*)d_in[8];
  const float* ln1_w       = (const float*)d_in[9];
  const float* ln1_b       = (const float*)d_in[10];
  const float* Wqkv        = (const float*)d_in[11];
  const float* bqkv        = (const float*)d_in[12];
  const float* Wo          = (const float*)d_in[13];
  const float* bo          = (const float*)d_in[14];
  const float* ln2_w       = (const float*)d_in[15];
  const float* ln2_b       = (const float*)d_in[16];
  const float* Wg          = (const float*)d_in[17];
  const float* bg          = (const float*)d_in[18];
  const float* W1          = (const float*)d_in[19];
  const float* b1          = (const float*)d_in[20];
  const float* W2          = (const float*)d_in[21];
  const float* b2          = (const float*)d_in[22];
  float* out = (float*)d_out;

  char* p = (char*)d_ws;
  float*    x      = (float*)p;    p += (size_t)MROWS * E * 4;
  _Float16* h16    = (_Float16*)p; p += (size_t)MROWS * E * 2;
  _Float16* qkv16  = (_Float16*)p; p += (size_t)MROWS * 3 * E * 2;
  _Float16* o16    = (_Float16*)p; p += (size_t)MROWS * E * 2;
  float*    gateb  = (float*)p;    p += (size_t)MROWS * NEXP * 4;
  _Float16* Wqkv16 = (_Float16*)p; p += (size_t)NL * 3 * E * E * 2;
  _Float16* Wo16   = (_Float16*)p; p += (size_t)NL * E * E * 2;
  _Float16* W1t    = (_Float16*)p; p += (size_t)NL * NEXP * E * FF * 2;
  _Float16* W2t    = (_Float16*)p; p += (size_t)NL * NEXP * FF * E * 2;

  {
    int total = NL * 3 * E * E + NL * E * E + 2 * NL * NEXP * E * FF;
    k_convert<<<(total + 255) / 256, 256, 0, stream>>>(Wqkv, Wo, W1, W2,
                                                       Wqkv16, Wo16, W1t, W2t);
  }
  k_embed_ln0<<<MROWS / 4, 128, 0, stream>>>(x_cat, x_num, x_eng, emb, emb_bias,
                                             emb_eng, emb_bias_en, ln0_w, ln0_b, x);
  for (int l = 0; l < NL; ++l) {
    k_ln_f16<<<MROWS / 4, 128, 0, stream>>>(x, ln1_w + l * E, ln1_b + l * E, h16);
    k_gemm_qkv<<<(MT / 8) * 6, 256, 0, stream>>>(h16, Wqkv16 + (size_t)l * 3 * E * E,
                                                 bqkv + l * 3 * E, qkv16);
    k_attn<<<B * H * 14, 128, 0, stream>>>(qkv16, o16);
    k_gemm_wo<<<(MT / 8) * 2, 256, 0, stream>>>(o16, Wo16 + (size_t)l * E * E,
                                                bo + l * E, x);
    k_ln_f16<<<MROWS / 4, 128, 0, stream>>>(x, ln2_w + l * E, ln2_b + l * E, h16);
    k_gate<<<(MROWS + 255) / 256, 256, 0, stream>>>(h16, Wg + (size_t)l * E * NEXP,
                                                    bg + l * NEXP, gateb);
    k_moe<<<MT, 256, 0, stream>>>(h16, W1t + (size_t)l * NEXP * FF * E,
                                  W2t + (size_t)l * NEXP * E * FF,
                                  b1 + l * NEXP * FF, b2 + l * NEXP * E, gateb, x);
  }
  k_out<<<(B * E + 255) / 256, 256, 0, stream>>>(x, out);
}
